// ResidualStack_20744692040345
// MI455X (gfx1250) — matrix operand order repaired and checked by reading, not compiled, not run on hardware
//
#include <hip/hip_runtime.h>
#include <hip/hip_bf16.h>

// CDNA5 / gfx1250 fused WaveNet residual-stack layer kernel.
// Each layer: Y = W0*X[:, :L-d] + W1*X[:, d:];  G = tanh(Y)*sigmoid(Y)
//             Res = Wr*G + X[:, d:]  (next layer input)
//             SkipAcc[tail window] += Ws*G
// All channel mixes run as v_wmma_f32_16x16x32_bf16 (f32 accumulate).

typedef __attribute__((ext_vector_type(16))) __bf16 v16bf;
typedef __attribute__((ext_vector_type(8)))  float  v8f;

#define TN  128   // time positions per workgroup tile
#define LDW 66    // padded LDS row stride in ushort (66*2B -> all-64-bank cycling)
#define NBLK 16
#define CCH 64
#define SCH 128

// Native bf16 convert (gfx1250 has hardware bf16 cvt; clang lowers the cast).
__device__ __forceinline__ unsigned short f2bf(float f) {
  union { __bf16 h; unsigned short s; } u;
  u.h = (__bf16)f;
  return u.s;
}

// A-fragment (16x32 bf16): lane m = lane&15; lanes<16 hold K 0-7 / 16-23,
// lanes>=16 hold K 8-15 / 24-31; 2 consecutive K per VGPR.
__device__ __forceinline__ v16bf load_A(const unsigned short* w, int mBase, int kBase, int lane) {
  union { unsigned int u[8]; v16bf v; } r;
  const unsigned short* row = w + (mBase + (lane & 15)) * LDW + kBase + ((lane & 16) ? 8 : 0);
#pragma unroll
  for (int i = 0; i < 8; ++i) {
    const int kk = ((i & 4) << 2) + ((i & 3) << 1);   // 0,2,4,6,16,18,20,22
    r.u[i] = *(const unsigned int*)(row + kk);
  }
  return r.v;
}

// B-fragment (32x16 bf16): lane n = lane&15; lanes<16 hold K 0-7 / 16-23,
// lanes>=16 hold K 8-15 / 24-31; 2 consecutive K per VGPR. LDS layout is [pos][ch].
__device__ __forceinline__ v16bf load_B(const unsigned short* x, int pBase, int cBase, int lane) {
  union { unsigned int u[8]; v16bf v; } r;
  const unsigned short* row = x + (pBase + (lane & 15)) * LDW + cBase + ((lane & 16) ? 8 : 0);
#pragma unroll
  for (int i = 0; i < 8; ++i) r.u[i] = *(const unsigned int*)(row + ((i & 4) << 2) + ((i & 3) << 1));
  return r.v;
}

__device__ __forceinline__ v8f wmma_bf16(v16bf a, v16bf b, v8f c) {
  return __builtin_amdgcn_wmma_f32_16x16x32_bf16(false, a, false, b, (short)0, c, false, false);
}

// g = tanh(y)*sigmoid(y) = (1-e^2) / ((1+e^2)(1+e)), e = exp(-y).
// One v_exp_f32 + one raw v_rcp_f32; clamp keeps e^2 finite (g==0 there anyway).
__device__ __forceinline__ float gate(float y) {
  const float yc = fmaxf(y, -30.0f);
  const float e1 = __expf(-yc);
  const float e2 = e1 * e1;
  return (1.0f - e2) * __builtin_amdgcn_rcpf((1.0f + e2) * (1.0f + e1));
}

__global__ void __launch_bounds__(256) wavenet_layer_kernel(
    const float* __restrict__ Xin, float* __restrict__ Xout,
    float* __restrict__ SkipAcc,
    const float* __restrict__ Wdil, const float* __restrict__ Wres,
    const float* __restrict__ Wskip,
    int d, int Lin, int Lout, int skipOff, int Lfin, int P /* = TN + d */)
{
  extern __shared__ unsigned short smem[];
  unsigned short* Xt  = smem;                 // P   x LDW  activations (bf16, [p][c])
  unsigned short* Gt  = Xt  + P   * LDW;      // TN  x LDW  gated activations
  unsigned short* W0l = Gt  + TN  * LDW;      // 64  x LDW  dil tap 0
  unsigned short* W1l = W0l + CCH * LDW;      // 64  x LDW  dil tap 1
  unsigned short* Wrl = W1l + CCH * LDW;      // 64  x LDW  residual 1x1
  unsigned short* Wsl = Wrl + CCH * LDW;      // 128 x LDW  skip 1x1

  const int tid  = threadIdx.x;
  const int lane = tid & 31;
  const int wv   = tid >> 5;                  // wave 0..7 -> position column
  const int n    = blockIdx.y;
  const int p0   = blockIdx.x * TN;

  // ---- stage weights (fp32 -> bf16) into LDS ----
  for (int e = tid; e < CCH * CCH; e += 256) {
    const int o = e >> 6, c = e & 63;
    W0l[o * LDW + c] = f2bf(Wdil[e * 2 + 0]);
    W1l[o * LDW + c] = f2bf(Wdil[e * 2 + 1]);
    Wrl[o * LDW + c] = f2bf(Wres[e]);
  }
  for (int e = tid; e < SCH * CCH; e += 256) {
    const int s = e >> 6, c = e & 63;
    Wsl[s * LDW + c] = f2bf(Wskip[e]);
  }

  // ---- stage activation tile [p][c] (covers left tap p0..p0+TN-1 and right tap +d) ----
  const float* Xn = Xin + (size_t)n * CCH * (size_t)Lin;
#pragma unroll 1
  for (int c = 0; c < CCH; ++c) {
    const float* col = Xn + (size_t)c * Lin;
    for (int p = tid; p < P; p += 256) {
      const int gp = p0 + p;
      Xt[p * LDW + c] = f2bf(gp < Lin ? col[gp] : 0.0f);
    }
  }
  __syncthreads();

  // ---- phase 1: dilated conv + gate -> Gt ----
  const int pB = wv * 16;
  const v16bf Bl0 = load_B(Xt, pB,     0,  lane);
  const v16bf Bl1 = load_B(Xt, pB,     32, lane);
  const v16bf Br0 = load_B(Xt, pB + d, 0,  lane);
  const v16bf Br1 = load_B(Xt, pB + d, 32, lane);

#pragma unroll
  for (int mt = 0; mt < 4; ++mt) {
    v8f acc = {};
    acc = wmma_bf16(load_A(W0l, mt * 16, 0,  lane), Bl0, acc);
    acc = wmma_bf16(load_A(W0l, mt * 16, 32, lane), Bl1, acc);
    acc = wmma_bf16(load_A(W1l, mt * 16, 0,  lane), Br0, acc);
    acc = wmma_bf16(load_A(W1l, mt * 16, 32, lane), Br1, acc);

    const int p  = pB + (lane & 15);
    const int cB = mt * 16 + ((lane & 16) ? 8 : 0);
    unsigned short gs[8];
#pragma unroll
    for (int v = 0; v < 8; ++v) gs[v] = f2bf(gate(acc[v]));
#pragma unroll
    for (int v = 0; v < 8; v += 2)
      *(unsigned int*)(&Gt[p * LDW + cB + v]) =
          (unsigned int)gs[v] | ((unsigned int)gs[v + 1] << 16);
  }
  __syncthreads();

  // ---- phase 2: residual and skip GEMMs over Gt ----
  const v16bf Bg0 = load_B(Gt, pB, 0,  lane);
  const v16bf Bg1 = load_B(Gt, pB, 32, lane);

  const int pg = p0 + pB + (lane & 15);
  float* Yn = Xout + (size_t)n * CCH * (size_t)Lout;

#pragma unroll
  for (int mt = 0; mt < 4; ++mt) {
    v8f acc = {};
    acc = wmma_bf16(load_A(Wrl, mt * 16, 0,  lane), Bg0, acc);
    acc = wmma_bf16(load_A(Wrl, mt * 16, 32, lane), Bg1, acc);
    if (pg < Lout) {
      const int cB = mt * 16 + ((lane & 16) ? 8 : 0);
#pragma unroll
      for (int v = 0; v < 8; ++v) {
        const int c = cB + v;
        Yn[(size_t)c * Lout + pg] = acc[v] + Xn[(size_t)c * Lin + pg + d];
      }
    }
  }

  float* Sn = SkipAcc + (size_t)n * SCH * (size_t)Lfin;
  const int j = pg - skipOff;   // tail-aligned final index
#pragma unroll
  for (int mt = 0; mt < 8; ++mt) {
    v8f acc = {};
    acc = wmma_bf16(load_A(Wsl, mt * 16, 0,  lane), Bg0, acc);
    acc = wmma_bf16(load_A(Wsl, mt * 16, 32, lane), Bg1, acc);
    if (pg < Lout && j >= 0) {
      const int sB = mt * 16 + ((lane & 16) ? 8 : 0);
#pragma unroll
      for (int v = 0; v < 8; ++v) {
        float* ptr = &Sn[(size_t)(sB + v) * Lfin + j];
        *ptr += acc[v];          // unique writer per element; layers are sequential
      }
    }
  }
}

__global__ void zero_f32_kernel(float* __restrict__ p, size_t nElem) {
  size_t i = (size_t)blockIdx.x * blockDim.x + threadIdx.x;
  const size_t stride = (size_t)gridDim.x * blockDim.x;
  for (; i < nElem; i += stride) p[i] = 0.0f;
}

extern "C" void kernel_launch(void* const* d_in, const int* in_sizes, int n_in,
                              void* d_out, int out_size, void* d_ws, size_t ws_size,
                              hipStream_t stream) {
  (void)in_sizes; (void)n_in; (void)out_size; (void)ws_size;
  const float* x  = (const float*)d_in[0];   // (16, 64, 8192)
  const float* Wd = (const float*)d_in[1];   // (16, 64, 64, 2)
  const float* Wr = (const float*)d_in[2];   // (16, 64, 64)
  const float* Ws = (const float*)d_in[3];   // (16, 128, 64)

  int dil[NBLK];
  { int k = 0; for (int s = 0; s < 2; ++s) for (int l = 0; l < 8; ++l) dil[k++] = 1 << l; }
  int Lfin = 8192;
  for (int i = 0; i < NBLK; ++i) Lfin -= dil[i];          // 7682

  float* outRes  = (float*)d_out;                          // (16, 64, 7682)
  float* skipOut = outRes + (size_t)16 * CCH * (size_t)Lfin; // (16, 128, 7682)

  // ping-pong activation buffers in workspace (fp32, inter-layer data L2-resident)
  float* bufA = (float*)d_ws;
  float* bufB = bufA + (size_t)16 * CCH * 8192;

  const int maxSmem =
      (int)(((size_t)(TN + 128) + TN + 3 * CCH + SCH) * LDW * sizeof(unsigned short));
  (void)hipFuncSetAttribute(reinterpret_cast<const void*>(wavenet_layer_kernel),
                            hipFuncAttributeMaxDynamicSharedMemorySize, maxSmem);

  zero_f32_kernel<<<1024, 256, 0, stream>>>(skipOut, (size_t)16 * SCH * (size_t)Lfin);

  int Lin = 8192;
  const float* cur = x;
  for (int i = 0; i < NBLK; ++i) {
    const int d    = dil[i];
    const int Lout = Lin - d;
    float* nxt = (i == NBLK - 1) ? outRes : ((i & 1) ? bufB : bufA);
    const int P = TN + d;
    const size_t smem = ((size_t)P + TN + 3 * CCH + SCH) * LDW * sizeof(unsigned short);
    dim3 grid((unsigned)((Lout + TN - 1) / TN), 16, 1);
    wavenet_layer_kernel<<<grid, 256, smem, stream>>>(
        cur, nxt, skipOut,
        Wd + (size_t)i * CCH * CCH * 2,
        Wr + (size_t)i * CCH * CCH,
        Ws + (size_t)i * SCH * CCH,
        d, Lin, Lout, Lout - Lfin, Lfin, P);
    cur = nxt;
    Lin = Lout;
  }
}
